// DeepGate3_structure_26611617366157
// MI455X (gfx1250) — matrix operand order repaired and checked by reading, not compiled, not run on hardware
//
#include <hip/hip_runtime.h>
#include <hip/hip_bf16.h>

// ---------------------------------------------------------------------------
// DeepGate3-structure forward for MI455X (gfx1250, wave32, WMMA bf16).
//   out = [hs | hf | pred_level | pred_connect | pred_hop_num]
// All GEMMs run on v_wmma_f32_16x16x32_bf16 with fragment-swizzled operands:
//   - weights pre-converted f32->bf16 and pre-swizzled into d_ws (one pass),
//     so a B fragment is one contiguous 32B load per lane (2x global_load_b128)
//   - LDS A operands stored in fragment order (2x ds_load_b128 per fragment)
// ---------------------------------------------------------------------------

typedef __attribute__((ext_vector_type(16))) __bf16 v16bf;
typedef __attribute__((ext_vector_type(8)))  float  v8f;

#define DI __device__ __forceinline__

DI int lane32() { return threadIdx.x & 31; }

DI v8f zero8() {
  v8f z;
#pragma unroll
  for (int i = 0; i < 8; ++i) z[i] = 0.f;
  return z;
}

DI v8f wmma_bf16(v16bf a, v16bf b, v8f c) {
  // D = A(16x32 bf16) * B(32x16 bf16) + C(16x16 f32)
  return __builtin_amdgcn_wmma_f32_16x16x32_bf16(false, a, false, b, (short)0, c,
                                                 false, false);
}

// Fragment-swizzled tile: 512 bf16, lane l owns elements [l*16 .. l*16+15].
// For element index i of lane l: m = l&15, k = 16*(i>>3) + 8*(l>>4) + (i&7).
// swz_idx inverts that: position of (m, kk) inside a 16x32 tile.
DI int swz_idx(int m, int kk) {
  int h = (kk >> 3) & 1;
  int l = m + 16 * h;
  int i = (kk & 7) | ((kk >> 4) << 3);
  return l * 16 + i;
}

// One fragment = this lane's 16 contiguous bf16 (32B) from a swizzled tile.
DI v16bf ld_frag(const __bf16* tile) {
  return *((const v16bf*)tile + lane32());
}

// ---------------------------------------------------------------------------
// Kernel 0: weight pre-swizzle  w[K][N] f32  ->  bf16 tiles (nt*KT + kt)*512
// ---------------------------------------------------------------------------
__global__ void swizzle_w_kernel(const float* __restrict__ w,
                                 __bf16* __restrict__ o, int K, int N) {
  int KT = K >> 5;
  int total = K * N;
  for (int idx = blockIdx.x * blockDim.x + threadIdx.x; idx < total;
       idx += gridDim.x * blockDim.x) {
    int t = idx >> 9, rem = idx & 511;
    int l = rem >> 4, i = rem & 15;
    int nt = t / KT, kt = t - nt * KT;
    int n = nt * 16 + (l & 15);
    int k = kt * 32 + 16 * (i >> 3) + 8 * (l >> 4) + (i & 7);
    o[idx] = (__bf16)w[(long)k * N + n];
  }
}

// ---------------------------------------------------------------------------
// Kernel 1: passthrough copy (float4, grid-stride)
// ---------------------------------------------------------------------------
__global__ void copy_f4_kernel(const float4* __restrict__ src,
                               float4* __restrict__ dst, long n) {
  long i = (long)blockIdx.x * blockDim.x + threadIdx.x;
  long stride = (long)gridDim.x * blockDim.x;
  for (; i < n; i += stride) dst[i] = src[i];
}

// ---------------------------------------------------------------------------
// Kernel 2: 3-layer MLP  relu(LN(x@W0+b0)) -> relu(LN(.@W1+b1)) -> .@W2+b2
// 128 threads (4 waves), 64 rows/block, wave owns one 16-row M-tile.
// GATHER builds cls_head pair features: feat f of row p =
//   hs[cpi[(f&1)*NP + p]][f>>1]
// ---------------------------------------------------------------------------
DI void ln_relu64_swz(float* __restrict__ H, __bf16* __restrict__ Oswz,
                      const float* __restrict__ g, const float* __restrict__ be) {
  for (int r = threadIdx.x; r < 64; r += 128) {
    float* row = H + r * 264;
    float mu = 0.f;
    for (int j = 0; j < 256; ++j) mu += row[j];
    mu *= (1.f / 256.f);
    float var = 0.f;
    for (int j = 0; j < 256; ++j) { float d = row[j] - mu; var += d * d; }
    var *= (1.f / 256.f);
    float rs = rsqrtf(var + 1e-5f);
    int mt = r >> 4, m = r & 15;
    for (int j = 0; j < 256; ++j) {
      float y = (row[j] - mu) * rs * g[j] + be[j];
      y = y > 0.f ? y : 0.f;
      row[j] = y;  // f32 post-activation (used by final dot layer)
      Oswz[(mt * 8 + (j >> 5)) * 512 + swz_idx(m, j & 31)] = (__bf16)y;
    }
  }
}

template <int DIN, bool GATHER>
__global__ __launch_bounds__(128) void mlp3_kernel(
    const float* __restrict__ X, const int* __restrict__ gidx, int gstride,
    const __bf16* __restrict__ W0s, const float* __restrict__ b0,
    const float* __restrict__ g0, const float* __restrict__ be0,
    const __bf16* __restrict__ W1s, const float* __restrict__ b1,
    const float* __restrict__ g1, const float* __restrict__ be1,
    const float* __restrict__ W2, const float* __restrict__ b2, int dout,
    float* __restrict__ out, int nrows) {
  const int KTA = DIN / 32;  // k-tiles of layer 0
  extern __shared__ char smem[];
  __bf16* aswz = (__bf16*)smem;                      // 4*KTA*512 bf16 swizzled
  float* h_f = (float*)(smem + (size_t)4 * KTA * 512 * sizeof(__bf16));
  __bf16* h_bf = (__bf16*)((char*)h_f + (size_t)64 * 264 * sizeof(float));
  // h_bf: 4 mtiles * 8 ktiles * 512 bf16 swizzled

  const int tid = threadIdx.x, wave = tid >> 5;
  const long row0 = (long)blockIdx.x * 64;

  // stage input tile as bf16, fragment-swizzled
  for (int idx = tid; idx < 64 * DIN; idx += 128) {
    int r = idx / DIN, f = idx - r * DIN;
    long row = row0 + r;
    float v = 0.f;
    if (row < nrows) {
      if (GATHER) {
        int node = gidx[(long)(f & 1) * gstride + row];
        v = X[(long)node * 128 + (f >> 1)];
      } else {
        v = X[row * (long)DIN + f];
      }
    }
    aswz[((r >> 4) * KTA + (f >> 5)) * 512 + swz_idx(r & 15, f & 31)] = (__bf16)v;
  }
  __syncthreads();

  // layer 0: [64,DIN] @ [DIN,256]
  {
    v16bf afr[KTA];
#pragma unroll
    for (int kt = 0; kt < KTA; ++kt)
      afr[kt] = ld_frag(aswz + (wave * KTA + kt) * 512);
    for (int nt = 0; nt < 16; ++nt) {
      v8f acc = zero8();
#pragma unroll
      for (int kt = 0; kt < KTA; ++kt)
        acc = wmma_bf16(afr[kt], ld_frag(W0s + (size_t)(nt * KTA + kt) * 512),
                        acc);
      int l = lane32(), n = nt * 16 + (l & 15), hh = l >> 4;
      float bv = b0[n];
#pragma unroll
      for (int r = 0; r < 8; ++r)
        h_f[(wave * 16 + r + 8 * hh) * 264 + n] = acc[r] + bv;
    }
  }
  __syncthreads();
  ln_relu64_swz(h_f, h_bf, g0, be0);
  __syncthreads();

  // layer 1: [64,256] @ [256,256]
  {
    v16bf afr[8];
#pragma unroll
    for (int kt = 0; kt < 8; ++kt)
      afr[kt] = ld_frag(h_bf + (wave * 8 + kt) * 512);
    for (int nt = 0; nt < 16; ++nt) {
      v8f acc = zero8();
#pragma unroll
      for (int kt = 0; kt < 8; ++kt)
        acc = wmma_bf16(afr[kt], ld_frag(W1s + (size_t)(nt * 8 + kt) * 512),
                        acc);
      int l = lane32(), n = nt * 16 + (l & 15), hh = l >> 4;
      float bv = b1[n];
#pragma unroll
      for (int r = 0; r < 8; ++r)
        h_f[(wave * 16 + r + 8 * hh) * 264 + n] = acc[r] + bv;
    }
  }
  __syncthreads();
  ln_relu64_swz(h_f, h_bf, g1, be1);
  __syncthreads();

  // layer 2: narrow output (dout in {1,3}) -> VALU dot per row (reads f32 h)
  for (int r = tid; r < 64; r += 128) {
    long row = row0 + r;
    if (row >= nrows) continue;
    for (int o = 0; o < dout; ++o) {
      float acc = b2[o];
      for (int j = 0; j < 256; ++j) acc += h_f[r * 264 + j] * W2[j * dout + o];
      out[row * dout + o] = acc;
    }
  }
}

// ---------------------------------------------------------------------------
// Kernel 3: batched hop transformer. 8 hops/block (80 rows = 5 exact M-tiles),
// 256 threads (8 waves), 3 post-norm encoder layers, float additive key mask.
// Writes CLS rows (f32 [4096,128]) to workspace for the readout_num MLP.
// ---------------------------------------------------------------------------
struct EncLayerP {
  const __bf16 *Wqkv, *Wo, *W1, *W2;  // fragment-swizzled bf16
  const float *bqkv, *bo, *ln1g, *ln1b, *b1f, *b2f, *ln2g, *ln2b;
};
struct EncP { EncLayerP L[3]; };

#define TF_XLD 132
#define TF_QLD 392

DI void tf_ln(float* __restrict__ xf, const float* __restrict__ tmpf,
              __bf16* __restrict__ xswz, const float* __restrict__ g,
              const float* __restrict__ b) {
  for (int r = threadIdx.x; r < 80; r += 256) {
    float mu = 0.f;
    for (int j = 0; j < 128; ++j) mu += xf[r * TF_XLD + j] + tmpf[r * TF_XLD + j];
    mu *= (1.f / 128.f);
    float var = 0.f;
    for (int j = 0; j < 128; ++j) {
      float d = xf[r * TF_XLD + j] + tmpf[r * TF_XLD + j] - mu;
      var += d * d;
    }
    var *= (1.f / 128.f);
    float rs = rsqrtf(var + 1e-5f);
    int mt = r >> 4, m = r & 15;
    for (int j = 0; j < 128; ++j) {
      float y = (xf[r * TF_XLD + j] + tmpf[r * TF_XLD + j] - mu) * rs * g[j] + b[j];
      xf[r * TF_XLD + j] = y;
      xswz[(mt * 4 + (j >> 5)) * 512 + swz_idx(m, j & 31)] = (__bf16)y;
    }
  }
}

__global__ __launch_bounds__(256) void hop_tf_kernel(
    const float* __restrict__ hs, const int* __restrict__ hop_pi,
    const int* __restrict__ stats, const int* __restrict__ hop_po,
    const float* __restrict__ cls_token, const float* __restrict__ pos_emb,
    EncP P, float* __restrict__ cls_out) {
  extern __shared__ char smem[];
  float* xf = (float*)smem;                                  // [80][132] f32
  __bf16* qkvb = (__bf16*)(smem + (size_t)80 * TF_XLD * 4);  // [80][392] bf16 linear
  float* tmpf = (float*)((char*)qkvb + (size_t)80 * TF_QLD * 2);  // [80][132] f32
  __bf16* obswz = (__bf16*)((char*)tmpf + (size_t)80 * TF_XLD * 4);  // 5*4*512 bf16
  __bf16* xswz = obswz + 5 * 4 * 512;                        // 5*4*512 bf16
  float* maskf = (float*)(xswz + 5 * 4 * 512);               // [8][16]
  __bf16* hswz = qkvb;  // FF chunk intermediate reuses qkv region (swizzled)

  const int tid = threadIdx.x, wave = tid >> 5;
  const int hop0 = blockIdx.x * 8;

  // ---- build embeddings: [cls | pi*valid | po] + pos_emb ----
  for (int idx = tid; idx < 80 * 128; idx += 256) {
    int r = idx >> 7, d = idx & 127;
    int hop = hop0 + r / 10, s = r % 10;
    float v;
    if (s == 0) {
      v = cls_token[d];
    } else if (s <= 8) {
      int slot = s - 1;
      int st = stats[hop * 8 + slot];
      int node = hop_pi[hop * 8 + slot];
      v = (st != -1) ? hs[(long)node * 128 + d] : 0.f;
    } else {
      int node = hop_po[hop];
      v = hs[(long)node * 128 + d];
    }
    v += pos_emb[s * 128 + d];
    xf[r * TF_XLD + d] = v;
    xswz[((r >> 4) * 4 + (d >> 5)) * 512 + swz_idx(r & 15, d & 31)] = (__bf16)v;
  }
  for (int idx = tid; idx < 8 * 16; idx += 256) {
    int h = idx >> 4, key = idx & 15;
    float m = 0.f;  // cls(0) and po(9) always valid -> +0
    if (key >= 1 && key <= 8)
      m = (stats[(hop0 + h) * 8 + (key - 1)] == -1) ? 1.f : 0.f;
    maskf[idx] = m;
  }
  __syncthreads();

  for (int l = 0; l < 3; ++l) {
    const EncLayerP& p = P.L[l];

    // ---- QKV: [80,128] @ [128,384] ----
    for (int mt = 0; mt < 5; ++mt) {
      v16bf afr[4];
#pragma unroll
      for (int kt = 0; kt < 4; ++kt) afr[kt] = ld_frag(xswz + (mt * 4 + kt) * 512);
      for (int nt = wave; nt < 24; nt += 8) {
        v8f acc = zero8();
#pragma unroll
        for (int kt = 0; kt < 4; ++kt)
          acc = wmma_bf16(afr[kt], ld_frag(p.Wqkv + (size_t)(nt * 4 + kt) * 512),
                          acc);
        int lp = lane32(), n = nt * 16 + (lp & 15), hh = lp >> 4;
        float bv = p.bqkv[n];
#pragma unroll
        for (int r = 0; r < 8; ++r)
          qkvb[(mt * 16 + r + 8 * hh) * TF_QLD + n] = (__bf16)(acc[r] + bv);
      }
    }
    __syncthreads();

    // ---- attention scores per (hop,head): Q(16x32) @ K^T(32x16) ----
    const float scale = 0.17677669529663687f;  // 1/sqrt(32)
    for (int inst = wave; inst < 32; inst += 8) {
      int hop = inst >> 2, head = inst & 3;
      int lp = lane32(), mq = lp & 15, hh = lp >> 4, nk = lp & 15;
      v16bf a, b;
#pragma unroll
      for (int i = 0; i < 16; ++i) {
        int k = 16 * (i >> 3) + 8 * hh + (i & 7);
        a[i] = (mq < 10) ? qkvb[(hop * 10 + mq) * TF_QLD + head * 32 + k]
                         : (__bf16)0.f;
      }
#pragma unroll
      for (int i = 0; i < 16; ++i) {
        int k = 16 * (i >> 3) + 8 * hh + (i & 7);  // B[k][n] = K[n][k]
        b[i] = (nk < 10) ? qkvb[(hop * 10 + nk) * TF_QLD + 128 + head * 32 + k]
                         : (__bf16)0.f;
      }
      v8f d = wmma_bf16(a, b, zero8());
      float madd = maskf[hop * 16 + nk];
#pragma unroll
      for (int r = 0; r < 8; ++r)
        tmpf[inst * 256 + (r + 8 * hh) * 16 + nk] = d[r] * scale + madd;
    }
    __syncthreads();

    // ---- softmax over the 10 keys ----
    for (int idx = tid; idx < 320; idx += 256) {
      int inst = idx / 10, q = idx % 10;
      float* s = tmpf + inst * 256 + q * 16;
      float mx = s[0];
      for (int k = 1; k < 10; ++k) mx = fmaxf(mx, s[k]);
      float e[10], sum = 0.f;
      for (int k = 0; k < 10; ++k) { e[k] = __expf(s[k] - mx); sum += e[k]; }
      float inv = 1.f / sum;
      for (int k = 0; k < 10; ++k) s[k] = e[k] * inv;
    }
    __syncthreads();

    // ---- O = attn(16x32, keys<10) @ V(32x16 x2) -> obswz (swizzled) ----
    for (int t = wave; t < 64; t += 8) {
      int inst = t >> 1, nt = t & 1;
      int hop = inst >> 2, head = inst & 3;
      int lp = lane32(), mq = lp & 15, hh = lp >> 4, nn = lp & 15;
      v16bf a, b;
#pragma unroll
      for (int i = 0; i < 16; ++i) {
        int k = 16 * (i >> 3) + 8 * hh + (i & 7);
        float v = (mq < 10 && k < 10) ? tmpf[inst * 256 + mq * 16 + k] : 0.f;
        a[i] = (__bf16)v;
      }
#pragma unroll
      for (int i = 0; i < 16; ++i) {
        int k = 16 * (i >> 3) + 8 * hh + (i & 7);
        b[i] = (k < 10) ? qkvb[(hop * 10 + k) * TF_QLD + 256 + head * 32 +
                               nt * 16 + nn]
                        : (__bf16)0.f;
      }
      v8f d = wmma_bf16(a, b, zero8());
#pragma unroll
      for (int r = 0; r < 8; ++r) {
        int m = r + 8 * hh;
        if (m < 10) {
          int row = hop * 10 + m;           // all 80 rows get written
          int col = head * 32 + nt * 16 + nn;
          obswz[((row >> 4) * 4 + (col >> 5)) * 512 +
                swz_idx(row & 15, col & 31)] = (__bf16)d[r];
        }
      }
    }
    __syncthreads();

    // ---- Wo: [80,128] @ [128,128] ----
    for (int mt = 0; mt < 5; ++mt) {
      v16bf afr[4];
#pragma unroll
      for (int kt = 0; kt < 4; ++kt)
        afr[kt] = ld_frag(obswz + (mt * 4 + kt) * 512);
      int nt = wave;  // 8 n-tiles, one per wave
      v8f acc = zero8();
#pragma unroll
      for (int kt = 0; kt < 4; ++kt)
        acc = wmma_bf16(afr[kt], ld_frag(p.Wo + (size_t)(nt * 4 + kt) * 512),
                        acc);
      int lp = lane32(), n = nt * 16 + (lp & 15), hh = lp >> 4;
      float bv = p.bo[n];
#pragma unroll
      for (int r = 0; r < 8; ++r)
        tmpf[(mt * 16 + r + 8 * hh) * TF_XLD + n] = acc[r] + bv;
    }
    __syncthreads();

    // ---- LN1: x = LN(x + attn_proj), also refresh xswz ----
    tf_ln(xf, tmpf, xswz, p.ln1g, p.ln1b);
    __syncthreads();

    // ---- FF fused: acc += relu(x@W1[:,c*128..]+b1) @ W2[c*128..,:] ----
    v8f facc[5];
#pragma unroll
    for (int i = 0; i < 5; ++i) facc[i] = zero8();
    for (int c = 0; c < 16; ++c) {
      // GEMM1 chunk -> hswz (swizzled bf16)
      for (int mt = 0; mt < 5; ++mt) {
        v16bf afr[4];
#pragma unroll
        for (int kt = 0; kt < 4; ++kt)
          afr[kt] = ld_frag(xswz + (mt * 4 + kt) * 512);
        int nt = wave;
        int ntg = c * 8 + nt;  // global n-tile of the 2048-wide hidden
        v8f acc = zero8();
#pragma unroll
        for (int kt = 0; kt < 4; ++kt)
          acc = wmma_bf16(afr[kt],
                          ld_frag(p.W1 + (size_t)(ntg * 4 + kt) * 512), acc);
        int lp = lane32(), nn = lp & 15, hh = lp >> 4;
        float bv = p.b1f[c * 128 + nt * 16 + nn];
#pragma unroll
        for (int r = 0; r < 8; ++r) {
          float y = acc[r] + bv;
          y = y > 0.f ? y : 0.f;
          int row = mt * 16 + r + 8 * hh;
          int col = nt * 16 + nn;
          hswz[((row >> 4) * 4 + (col >> 5)) * 512 +
               swz_idx(row & 15, col & 31)] = (__bf16)y;
        }
      }
      __syncthreads();
      // GEMM2 accumulate: wave owns n-tile = wave, all 5 m-tiles
#pragma unroll
      for (int mt = 0; mt < 5; ++mt) {
        v16bf afr[4];
#pragma unroll
        for (int kt = 0; kt < 4; ++kt)
          afr[kt] = ld_frag(hswz + (mt * 4 + kt) * 512);
        int nt = wave;
#pragma unroll
        for (int kt = 0; kt < 4; ++kt)
          facc[mt] = wmma_bf16(
              afr[kt],
              ld_frag(p.W2 + (size_t)(nt * 64 + c * 4 + kt) * 512), facc[mt]);
      }
      __syncthreads();
    }
#pragma unroll
    for (int mt = 0; mt < 5; ++mt) {
      int lp = lane32(), n = wave * 16 + (lp & 15), hh = lp >> 4;
      float bv = p.b2f[n];
#pragma unroll
      for (int r = 0; r < 8; ++r)
        tmpf[(mt * 16 + r + 8 * hh) * TF_XLD + n] = facc[mt][r] + bv;
    }
    __syncthreads();

    // ---- LN2: x = LN(x + ff), refresh xswz ----
    tf_ln(xf, tmpf, xswz, p.ln2g, p.ln2b);
    __syncthreads();
  }

  // ---- export CLS rows ----
  for (int idx = tid; idx < 8 * 128; idx += 256) {
    int h = idx >> 7, d = idx & 127;
    cls_out[(long)(hop0 + h) * 128 + d] = xf[(h * 10) * TF_XLD + d];
  }
}

// ---------------------------------------------------------------------------
// Host launcher
// ---------------------------------------------------------------------------
extern "C" void kernel_launch(void* const* d_in, const int* in_sizes, int n_in,
                              void* d_out, int out_size, void* d_ws,
                              size_t ws_size, hipStream_t stream) {
  auto F = [&](int i) { return (const float*)d_in[i]; };

  // top-level inputs (setup_inputs insertion order)
  const float* hs = F(0);
  const float* hf = F(1);
  const int* cpi = (const int*)d_in[2];        // [2, 100000]
  const int* hop_pi = (const int*)d_in[3];     // [4096, 8]
  const int* hop_stats = (const int*)d_in[4];  // [4096, 8]
  const int* hop_po = (const int*)d_in[5];     // [4096, 1]

  // params: jax pytree flatten (dict keys sorted lexicographically)
  const float *ch_W0 = F(6), *ch_W1 = F(7), *ch_W2 = F(8);
  const float *ch_b0 = F(9), *ch_b1 = F(10), *ch_b2 = F(11);
  const float *ch_be0 = F(12), *ch_be1 = F(13), *ch_g0 = F(14), *ch_g1 = F(15);
  const float* cls_token = F(16);
  const float* pos_emb = F(17);
  const float *rl_W0 = F(18), *rl_W1 = F(19), *rl_W2 = F(20);
  const float *rl_b0 = F(21), *rl_b1 = F(22), *rl_b2 = F(23);
  const float *rl_be0 = F(24), *rl_be1 = F(25), *rl_g0 = F(26), *rl_g1 = F(27);
  const float *rn_W0 = F(28), *rn_W1 = F(29), *rn_W2 = F(30);
  const float *rn_b0 = F(31), *rn_b1 = F(32), *rn_b2 = F(33);
  const float *rn_be0 = F(34), *rn_be1 = F(35), *rn_g0 = F(36), *rn_g1 = F(37);

  // workspace layout: [0,2MB) f32 CLS rows; [2MB,6MB) swizzled bf16 weights
  float* cls_ws = (float*)d_ws;
  __bf16* wsw = (__bf16*)((char*)d_ws + (size_t)2 * 1024 * 1024);

  // swizzled weight element offsets
  const size_t o_rlW0 = 0, o_rlW1 = 32768;
  const size_t o_chW0 = 98304, o_chW1 = 163840;
  const size_t o_rnW0 = 229376, o_rnW1 = 262144;
  const size_t o_tf = 327680, tf_sz = 589824;  // per layer
  const size_t o_qkv = 0, o_wo = 49152, o_w1 = 65536, o_w2 = 327680;

  struct SwzJob { const float* w; size_t off; int K, N; };
  SwzJob jobs[18] = {
      {rl_W0, o_rlW0, 128, 256}, {rl_W1, o_rlW1, 256, 256},
      {ch_W0, o_chW0, 256, 256}, {ch_W1, o_chW1, 256, 256},
      {rn_W0, o_rnW0, 128, 256}, {rn_W1, o_rnW1, 256, 256},
      {nullptr, 0, 0, 0}, {nullptr, 0, 0, 0}, {nullptr, 0, 0, 0},
      {nullptr, 0, 0, 0}, {nullptr, 0, 0, 0}, {nullptr, 0, 0, 0},
      {nullptr, 0, 0, 0}, {nullptr, 0, 0, 0}, {nullptr, 0, 0, 0},
      {nullptr, 0, 0, 0}, {nullptr, 0, 0, 0}, {nullptr, 0, 0, 0}};

  EncP ep;
  for (int l = 0; l < 3; ++l) {
    int b = 38 + l * 12;  // sorted: W1 W2 Wo Wqkv b1f b2f bo bqkv ln1_b ln1_g ln2_b ln2_g
    size_t base = o_tf + (size_t)l * tf_sz;
    jobs[6 + l * 4 + 0] = {F(b + 3), base + o_qkv, 128, 384};   // Wqkv
    jobs[6 + l * 4 + 1] = {F(b + 2), base + o_wo, 128, 128};    // Wo
    jobs[6 + l * 4 + 2] = {F(b + 0), base + o_w1, 128, 2048};   // W1
    jobs[6 + l * 4 + 3] = {F(b + 1), base + o_w2, 2048, 128};   // W2
    ep.L[l].Wqkv = wsw + base + o_qkv;
    ep.L[l].Wo = wsw + base + o_wo;
    ep.L[l].W1 = wsw + base + o_w1;
    ep.L[l].W2 = wsw + base + o_w2;
    ep.L[l].b1f = F(b + 4);  ep.L[l].b2f = F(b + 5);
    ep.L[l].bo = F(b + 6);   ep.L[l].bqkv = F(b + 7);
    ep.L[l].ln1b = F(b + 8); ep.L[l].ln1g = F(b + 9);
    ep.L[l].ln2b = F(b + 10); ep.L[l].ln2g = F(b + 11);
  }

  // 0) weight pre-swizzle (f32 -> fragment-ordered bf16)
  for (int j = 0; j < 18; ++j) {
    int total = jobs[j].K * jobs[j].N;
    int blocks = (total + 255) / 256;
    if (blocks > 512) blocks = 512;
    swizzle_w_kernel<<<blocks, 256, 0, stream>>>(jobs[j].w, wsw + jobs[j].off,
                                                 jobs[j].K, jobs[j].N);
  }

  float* out = (float*)d_out;
  const long O_HF = 25600000L;
  const long O_LEVEL = 51200000L;
  const long O_CONN = 51400000L;
  const long O_NUM = 51700000L;

  // dynamic LDS sizes
  const size_t SM128 = (size_t)4 * 4 * 512 * 2 + (size_t)64 * 264 * 4 +
                       (size_t)4 * 8 * 512 * 2;  // 116736
  const size_t SM256 = (size_t)4 * 8 * 512 * 2 + (size_t)64 * 264 * 4 +
                       (size_t)4 * 8 * 512 * 2;  // 133120
  const size_t SMTF = (size_t)80 * TF_XLD * 4 + (size_t)80 * TF_QLD * 2 +
                      (size_t)80 * TF_XLD * 4 + (size_t)5 * 4 * 512 * 2 * 2 +
                      (size_t)8 * 16 * 4;  // 188672

  // 1) passthrough copies (bandwidth floor)
  copy_f4_kernel<<<2048, 256, 0, stream>>>((const float4*)hs, (float4*)out,
                                           6400000L);
  copy_f4_kernel<<<2048, 256, 0, stream>>>((const float4*)hf,
                                           (float4*)(out + O_HF), 6400000L);

  // 2) readout_level on 200000 nodes
  mlp3_kernel<128, false><<<3125, 128, SM128, stream>>>(
      hs, nullptr, 0, wsw + o_rlW0, rl_b0, rl_g0, rl_be0, wsw + o_rlW1, rl_b1,
      rl_g1, rl_be1, rl_W2, rl_b2, 1, out + O_LEVEL, 200000);

  // 3) cls_head on 100000 gathered pair features
  mlp3_kernel<256, true><<<1563, 128, SM256, stream>>>(
      hs, cpi, 100000, wsw + o_chW0, ch_b0, ch_g0, ch_be0, wsw + o_chW1, ch_b1,
      ch_g1, ch_be1, ch_W2, ch_b2, 3, out + O_CONN, 100000);

  // 4) hop transformer (8 hops/block) -> CLS rows to workspace
  hop_tf_kernel<<<512, 256, SMTF, stream>>>(hs, hop_pi, hop_stats, hop_po,
                                            cls_token, pos_emb, ep, cls_ws);

  // 5) readout_num on 4096 CLS rows
  mlp3_kernel<128, false><<<64, 128, SM128, stream>>>(
      cls_ws, nullptr, 0, wsw + o_rnW0, rn_b0, rn_g0, rn_be0, wsw + o_rnW1,
      rn_b1, rn_g1, rn_be1, rn_W2, rn_b2, 1, out + O_NUM, 4096);
}
